// Generator_29678224015855
// MI455X (gfx1250) — matrix operand order repaired and checked by reading, not compiled, not run on hardware
//
#include <hip/hip_runtime.h>
#include <hip/hip_bf16.h>
#include <stdint.h>

// Problem constants (match reference)
#define TSEQ 239
#define BATCH 4096
#define HID 96
#define NLAYERS 4

typedef __attribute__((ext_vector_type(16))) __bf16 v16bf;
typedef __attribute__((ext_vector_type(8)))  float  v8f;

// TDM descriptor vector types (probe-verified builtin signature)
typedef __attribute__((ext_vector_type(4))) unsigned tdm_v4u;
typedef __attribute__((ext_vector_type(8))) int      tdm_v8i;
typedef __attribute__((ext_vector_type(4))) int      tdm_v4i;

union Frag { v16bf v; uint4 q[2]; __bf16 e[16]; };

__device__ inline v8f wmma_bf16(v16bf a, v16bf b, v8f c) {
  // D = A(16x32 bf16) * B(32x16 bf16) + C(16x16 f32)
  return __builtin_amdgcn_wmma_f32_16x16x32_bf16(false, a, false, b, (short)0, c,
                                                 false, false);
}

// A-operand fragment (16x32 bf16 tile), source row-major with row stride ld.
// ISA layout: lanes 0-15 hold K = k0+0..7 (v0-3) and k0+16..23 (v4-7);
// lanes 16-31 hold K = k0+8..15 and k0+24..31. Row M = lane&15.
__device__ inline v16bf load_a_frag(const __bf16* base, int ld, int m, int k0, int half) {
  Frag f;
  const __bf16* p = base + (size_t)m * ld + k0 + half * 8;
  f.q[0] = *(const uint4*)(p);
  f.q[1] = *(const uint4*)(p + 16);
  return f.v;
}

// B-operand fragment (32x16 bf16, K x N) sourced from W[n][k] row-major stride ld.
// ISA layout: lanes 0-15 hold K = k0+0..7 and k0+16..23, lanes 16-31 K = k0+8..15 and k0+24..31; N = lane&15.
__device__ inline v16bf load_b_frag(const __bf16* base, int ld, int n, int k0, int half) {
  Frag f;
  const __bf16* p = base + (size_t)n * ld + k0 + half * 8;
  f.q[0] = *(const uint4*)(p);
  f.q[1] = *(const uint4*)(p + 16);
  return f.v;
}

__device__ inline float sigm(float x) { return 1.0f / (1.0f + __expf(-x)); }
__device__ inline float tanh_fast(float x) {
  x = fminf(fmaxf(x, -15.0f), 15.0f);
  float e = __expf(2.0f * x);
  return (e - 1.0f) / (e + 1.0f);
}

// ---------------------------------------------------------------------------
// TDM: async DMA of one contiguous run of `n2` bf16 elements global -> LDS.
// Builds the D# per CDNA5 ISA 8.3/8.4: 1D tile (tile_dim0=n2), data_size=2B,
// type=2 ("image"), no multicast / padding / iteration. Tracked by TENSORcnt.
// ---------------------------------------------------------------------------
__device__ inline void tdm_load_contig_bf16(unsigned lds_off, const void* gsrc,
                                            unsigned n2) {
  unsigned long long ga = (unsigned long long)(uintptr_t)gsrc;
  tdm_v4u g0 = {
      1u,                                   // count=1, user descriptor
      lds_off,                              // lds_addr (bytes)
      (unsigned)ga,                         // global_addr[31:0]
      (unsigned)((ga >> 32) & 0x01FFFFFFu)  // global_addr[56:32]
          | (2u << 30)};                    // type=2
  tdm_v8i g1 = {
      (int)(1u << 16),                      // data_size=1 (2 bytes), mask=0
      (int)((n2 & 0xFFFFu) << 16),          // tensor_dim0[15:0]
      (int)((n2 >> 16) | (1u << 16)),       // tensor_dim0[31:16] | tensor_dim1=1
      (int)((n2 & 0xFFFFu) << 16),          // tile_dim0
      1,                                    // tile_dim1=1, tile_dim2=0
      (int)n2,                              // tensor_dim0_stride[31:0]
      0, 0};                                // stride hi / tensor_dim1_stride
  tdm_v4i zz = {0, 0, 0, 0};
#if __clang_major__ >= 23
  tdm_v8i z8 = {0, 0, 0, 0, 0, 0, 0, 0};
  __builtin_amdgcn_tensor_load_to_lds(g0, g1, zz, zz, z8, 0);
#else
  __builtin_amdgcn_tensor_load_to_lds(g0, g1, zz, zz, 0);
#endif
}

// ---------------------------------------------------------------------------
// Weight fp32 -> bf16 conversion (one-time)
// ---------------------------------------------------------------------------
__global__ void convert_weights_kernel(const float* __restrict__ wih_f,
                                       const float* __restrict__ whh_f,
                                       __bf16* __restrict__ wih_b,
                                       __bf16* __restrict__ whh_b, int n) {
  int i = blockIdx.x * blockDim.x + threadIdx.x;
  if (i < n) {
    wih_b[i] = (__bf16)wih_f[i];
    whh_b[i] = (__bf16)whh_f[i];
  }
}

// ---------------------------------------------------------------------------
// fc1: [4096,512] @ [512,384]^T + bias, LeakyReLU(0.2) -> hbuf [4096,384] f32
// ---------------------------------------------------------------------------
__global__ __launch_bounds__(128) void fc1_kernel(const float* __restrict__ z,
                                                  const float* __restrict__ c,
                                                  const float* __restrict__ w,
                                                  const float* __restrict__ bias,
                                                  float* __restrict__ hbuf) {
  const int tid = threadIdx.x, lane = tid & 31, wv = tid >> 5;
  const int wid = blockIdx.x * 4 + wv;
  const int mt = wid / 24, nt = wid % 24;  // 256 x 24 tiles
  const int lm = lane & 15, half = lane >> 4;
  const int mrow = mt * 16 + lm;
  const int nrow = nt * 16 + lm;

  v8f acc;
#pragma unroll
  for (int v = 0; v < 8; ++v) acc[v] = 0.0f;

#pragma unroll 4
  for (int kc = 0; kc < 16; ++kc) {
    const int k0 = kc * 32;
    const float* ap = (k0 < 256) ? (z + (size_t)mrow * 256 + k0)
                                 : (c + (size_t)mrow * 256 + (k0 - 256));
    Frag a, b;
#pragma unroll
    for (int i = 0; i < 8; ++i) a.e[i] = (__bf16)ap[half * 8 + i];
#pragma unroll
    for (int i = 0; i < 8; ++i) a.e[8 + i] = (__bf16)ap[16 + half * 8 + i];
    const float* bp = w + (size_t)nrow * 512 + k0 + half * 8;
#pragma unroll
    for (int i = 0; i < 16; ++i) b.e[i] = (__bf16)bp[(i < 8) ? i : (i + 8)];
    acc = wmma_bf16(a.v, b.v, acc);
  }

  const float bv = bias[nrow];
#pragma unroll
  for (int v = 0; v < 8; ++v) {
    float x = acc[v] + bv;
    x = (x >= 0.0f) ? x : 0.2f * x;  // LeakyReLU(0.2)
    hbuf[(size_t)(mt * 16 + v + 8 * half) * 384 + nrow] = x;
  }
}

// ---------------------------------------------------------------------------
// One GRU layer. 256 WGs x 192 threads (6 waves). WG owns 16 batch rows; wave
// jg owns columns [16jg,16jg+16) of all three gates (gate math is lane-local).
//  - recurrent weight B-frags: registers (reused 239x, serial critical path)
//  - input weights: staged once into padded LDS (conflict-free ds_load_b128)
//  - x_t tiles: TDM DMA into double-buffered LDS, issued 2 steps ahead,
//    fenced with s_wait_tensorcnt before the per-step barrier
// ---------------------------------------------------------------------------
__global__ __launch_bounds__(192) void gru_layer_kernel(
    const __bf16* __restrict__ x_in, __bf16* __restrict__ y_bf,
    float* __restrict__ y_f32, const __bf16* __restrict__ wih,
    const __bf16* __restrict__ whh, const float* __restrict__ bih,
    const float* __restrict__ bhh, const float* __restrict__ hbuf, int layer) {
  __shared__ __attribute__((aligned(16))) __bf16 wil[288 * 104];   // padded wi
  __shared__ __attribute__((aligned(16))) __bf16 xsh[2][16][HID];  // TDM dest
  __shared__ __attribute__((aligned(16))) __bf16 hsh[2][16][104];  // h hand-off

  const int tid = threadIdx.x;
  const int lane = tid & 31;
  const int jg = tid >> 5;  // wave id == j-tile (0..5)
  const int lm = lane & 15;
  const int half = lane >> 4;
  const int row0 = blockIdx.x * 16;
  const int j = jg * 16 + lm;  // hidden column of this lane's C tile

  const float bR = bih[j] + bhh[j];
  const float bZ = bih[96 + j] + bhh[96 + j];
  const float bIN = bih[192 + j];
  const float bHN = bhh[192 + j];

  // Recurrent weight fragments in registers for the whole time loop.
  v16bf bh[3][3];
#pragma unroll
  for (int g = 0; g < 3; ++g)
#pragma unroll
    for (int kc = 0; kc < 3; ++kc)
      bh[g][kc] = load_b_frag(whh, HID, g * 96 + j, kc * 32, half);

  if (x_in) {
    // Stage input weights into padded LDS (row stride 104 -> bank-clean).
    for (int idx = tid; idx < 288 * 12; idx += 192) {
      const int r = idx / 12, cb = idx % 12;
      *(uint4*)&wil[r * 104 + cb * 8] = *(const uint4*)(wih + r * 96 + cb * 8);
    }
    if (jg == 0) {  // TDM prologue: x_0 (wait), x_1 (in flight)
      tdm_load_contig_bf16((unsigned)(uintptr_t)&xsh[0][0][0],
                           x_in + (size_t)row0 * HID, 16 * HID);
      __builtin_amdgcn_s_wait_tensorcnt(0);
      tdm_load_contig_bf16((unsigned)(uintptr_t)&xsh[1][0][0],
                           x_in + ((size_t)BATCH + row0) * HID, 16 * HID);
    }
  }

  // h0 from fc1 output: h0[l,b,j] = hbuf[l*1024 + b/4][(b%4)*96 + j]
  float hp[8];
#pragma unroll
  for (int v = 0; v < 8; ++v) {
    const int b = row0 + v + 8 * half;
    hp[v] = hbuf[(size_t)(layer * 1024 + (b >> 2)) * 384 + (b & 3) * 96 + j];
    hsh[0][v + 8 * half][j] = (__bf16)hp[v];
  }
  __syncthreads();

  for (int t = 0; t < TSEQ; ++t) {
    const int cur = t & 1, nxt = cur ^ 1;

    v16bf ah[3];
#pragma unroll
    for (int kc = 0; kc < 3; ++kc)
      ah[kc] = load_a_frag(&hsh[cur][0][0], 104, lm, kc * 32, half);

    v8f aR, aZ, aIN, aHN;
#pragma unroll
    for (int v = 0; v < 8; ++v) { aR[v] = bR; aZ[v] = bZ; aIN[v] = bIN; aHN[v] = bHN; }

    // Recurrent GEMM: gh = h @ wh^T
#pragma unroll
    for (int kc = 0; kc < 3; ++kc) {
      aR = wmma_bf16(ah[kc], bh[0][kc], aR);
      aZ = wmma_bf16(ah[kc], bh[1][kc], aZ);
      aHN = wmma_bf16(ah[kc], bh[2][kc], aHN);
    }

    // Input GEMM: gi = x_t @ wi^T (layer 0 input is zeros -> bias only)
    if (x_in) {
      const __bf16* xb = &xsh[cur][0][0];
#pragma unroll
      for (int kc = 0; kc < 3; ++kc) {
        v16bf ax = load_a_frag(xb, HID, lm, kc * 32, half);
        aR = wmma_bf16(ax, load_b_frag(wil, 104, j, kc * 32, half), aR);
        aZ = wmma_bf16(ax, load_b_frag(wil, 104, 96 + j, kc * 32, half), aZ);
        aIN = wmma_bf16(ax, load_b_frag(wil, 104, 192 + j, kc * 32, half), aIN);
      }
    }

    // Gate math (lane-local: j and b match the owned h element).
#pragma unroll
    for (int v = 0; v < 8; ++v) {
      const float r = sigm(aR[v]);
      const float z = sigm(aZ[v]);
      const float n = tanh_fast(aIN[v] + r * aHN[v]);
      const float h = (1.0f - z) * n + z * hp[v];
      hp[v] = h;
      hsh[nxt][v + 8 * half][j] = (__bf16)h;
      if (y_f32)  // final layer: transposed [B,H,T] fp32 from registers
        y_f32[(size_t)(row0 + v + 8 * half) * (HID * TSEQ) + (size_t)j * TSEQ + t] = h;
    }

    // x_{t+1} must be resident in xsh[nxt] before anyone passes the barrier.
    if (x_in && jg == 0) __builtin_amdgcn_s_wait_tensorcnt(0);
    __syncthreads();  // one barrier per step (double-buffered h and x)

    // Coalesced bf16 sequence store for layers 0..2.
    if (y_bf) {
      const int r_ = tid / 12, cc = tid % 12;  // 16 rows x 12 chunks of 8
      uint4 d = *(const uint4*)&hsh[nxt][r_][cc * 8];
      *(uint4*)(y_bf + ((size_t)t * BATCH + row0 + r_) * HID + cc * 8) = d;
    }

    // Kick the DMA for x_{t+2} into the buffer just released (xsh[cur]).
    if (x_in && jg == 0 && t + 2 < TSEQ)
      tdm_load_contig_bf16((unsigned)(uintptr_t)&xsh[cur][0][0],
                           x_in + ((size_t)(t + 2) * BATCH + row0) * HID, 16 * HID);
  }
}

// ---------------------------------------------------------------------------
// Host-side orchestration. Workspace (~195 MB):
//   seqA bf16 [T,B,96] | hbuf f32 [4096,384] | wih_b | whh_b (bf16)
// d_out (376 MB fp32) doubles as the bf16 pong buffer for layer 1's sequence;
// layer 3 overwrites all of d_out with the final fp32 [B,H,T] result.
// ---------------------------------------------------------------------------
extern "C" void kernel_launch(void* const* d_in, const int* in_sizes, int n_in,
                              void* d_out, int out_size, void* d_ws, size_t ws_size,
                              hipStream_t stream) {
  const float* z = (const float*)d_in[0];
  const float* c = (const float*)d_in[1];
  const float* fc1_w = (const float*)d_in[2];
  const float* fc1_b = (const float*)d_in[3];
  const float* w_ih = (const float*)d_in[4];
  const float* w_hh = (const float*)d_in[5];
  const float* b_ih = (const float*)d_in[6];
  const float* b_hh = (const float*)d_in[7];

  const size_t seq_elems = (size_t)TSEQ * BATCH * HID;
  char* ws = (char*)d_ws;
  __bf16* seqA = (__bf16*)ws;
  float* hbuf = (float*)(ws + seq_elems * 2);
  __bf16* wih_b = (__bf16*)(ws + seq_elems * 2 + (size_t)BATCH * 384 * 4);
  __bf16* whh_b = wih_b + (size_t)NLAYERS * 288 * HID;
  __bf16* seqB = (__bf16*)d_out;  // scratch region inside the output buffer

  const int wn = NLAYERS * 288 * HID;
  convert_weights_kernel<<<(wn + 255) / 256, 256, 0, stream>>>(w_ih, w_hh, wih_b,
                                                               whh_b, wn);
  fc1_kernel<<<(256 * 24) / 4, 128, 0, stream>>>(z, c, fc1_w, fc1_b, hbuf);

  dim3 grid(BATCH / 16), blk(192);
  gru_layer_kernel<<<grid, blk, 0, stream>>>(nullptr, seqA, nullptr, wih_b, whh_b,
                                             b_ih, b_hh, hbuf, 0);
  gru_layer_kernel<<<grid, blk, 0, stream>>>(seqA, seqB, nullptr,
                                             wih_b + 1 * 288 * HID,
                                             whh_b + 1 * 288 * HID, b_ih + 288,
                                             b_hh + 288, hbuf, 1);
  gru_layer_kernel<<<grid, blk, 0, stream>>>(seqB, seqA, nullptr,
                                             wih_b + 2 * 288 * HID,
                                             whh_b + 2 * 288 * HID, b_ih + 2 * 288,
                                             b_hh + 2 * 288, hbuf, 2);
  gru_layer_kernel<<<grid, blk, 0, stream>>>(seqA, nullptr, (float*)d_out,
                                             wih_b + 3 * 288 * HID,
                                             whh_b + 3 * 288 * HID, b_ih + 3 * 288,
                                             b_hh + 3 * 288, hbuf, 3);
}
